// CharRNN_18176301597378
// MI455X (gfx1250) — matrix operand order repaired and checked by reading, not compiled, not run on hardware
//
#include <hip/hip_runtime.h>
#include <hip/hip_bf16.h>
#include <math.h>

// Problem constants (match reference)
#define VSZ 50000
#define BSZ 128
#define SSZ 256
#define USZ 512
#define NSS 64

typedef _Float16 v16h __attribute__((ext_vector_type(16)));
typedef float    v8f  __attribute__((ext_vector_type(8)));

// ---------------------------------------------------------------------------
// WMMA fragment helpers (wave32, v_wmma_f32_16x16x32_f16)
//
// A (16x32 f16): lane half = lane>>4, m = lane&15.
//   a[0..15] = A[m][kofs + half*16 + 0..15]
// B (32x16 f16): lane n = lane&15, half = lane>>4.
//   b[0..15] = B[half*16 + 0..15][n]  -> pre-packed so each lane reads 32
//   contiguous bytes; a wave reads one contiguous 1KB fragment block.
// C/D (16x16 f32): c[i] is element (M = half*8 + i, N = lane&15).
// ---------------------------------------------------------------------------

__device__ __forceinline__ v16h load_a_f16(const _Float16* __restrict__ row, int kofs, int half) {
  const _Float16* p = row + kofs + half * 16;
  v16h a;
#pragma unroll
  for (int j = 0; j < 16; ++j) a[j] = p[j];
  return a;
}

__device__ __forceinline__ v16h load_b_frag(const _Float16* __restrict__ base) {
  v16h b;
#pragma unroll
  for (int j = 0; j < 16; ++j) b[j] = base[j];
  return b;
}

__device__ __forceinline__ v8f wmma_f16(v16h a, v16h b, v8f c) {
  // (neg_a, A, neg_b, B, c_mod, C, reuse_a, reuse_b)
  return __builtin_amdgcn_wmma_f32_16x16x32_f16(false, a, false, b, (short)0, c, false, false);
}

__device__ __forceinline__ float softplusf(float x) {
  return fmaxf(x, 0.0f) + log1pf(__expf(-fabsf(x)));
}

__device__ __forceinline__ float log_q_dev(int c) {
  // log((log(c+2)-log(c+1))/log(V+1)); log(c+2)-log(c+1) == log1p(1/(c+1))
  float cf = (float)c;
  float d = log1pf(1.0f / (cf + 1.0f));
  return __logf(d) - __logf((float)(VSZ + 1));
}

// ---------------------------------------------------------------------------
// One-time packing / gather kernels (parallel, off the serial critical path)
// ---------------------------------------------------------------------------

// Pack row-major f32 W[Kdim][N] into fragment-major f16 B blocks:
// dst[(((ktile*NT)+ntile)*32 + lane)*16 + j] = W[ktile*32 + half*16 + j][ntile*16 + n]
__global__ void pack_weight(const float* __restrict__ W, _Float16* __restrict__ dst,
                            int Kdim, int N) {
  int idx = blockIdx.x * blockDim.x + threadIdx.x;
  if (idx >= Kdim * N) return;
  int j    = idx & 15;
  int lane = (idx >> 4) & 31;
  int NT   = N >> 4;
  int rest = idx >> 9;              // (ktile*NT + ntile), since 32*16 = 512
  int ntile = rest % NT;
  int ktile = rest / NT;
  int n = lane & 15, half = lane >> 4;
  int K = ktile * 32 + half * 16 + j;
  dst[idx] = (_Float16)W[(size_t)K * N + ntile * 16 + n];
}

// Pack sampled softmax rows: B[k][n] = softmax_w[k*V + sampled_ids[n]], K=512, N=64
__global__ void pack_wns(const float* __restrict__ sw, const int* __restrict__ sids,
                         _Float16* __restrict__ dst) {
  int idx = blockIdx.x * blockDim.x + threadIdx.x;  // 512*64
  if (idx >= USZ * NSS) return;
  int j    = idx & 15;
  int lane = (idx >> 4) & 31;
  const int NT = NSS >> 4;          // 4
  int rest = idx >> 9;
  int ntile = rest % NT;
  int ktile = rest / NT;
  int n = lane & 15, half = lane >> 4;
  int K = ktile * 32 + half * 16 + j;
  dst[idx] = (_Float16)sw[(size_t)K * VSZ + sids[ntile * 16 + n]];
}

__global__ void colbias_k(const float* __restrict__ sb, const int* __restrict__ sids,
                          float* __restrict__ cbias) {
  int n = threadIdx.x;
  if (n >= NSS) return;
  int sid = sids[n];
  cbias[n] = sb[sid] - log_q_dev(sid);
}

__global__ void zero_f32(float* __restrict__ p, int n) {
  int i = blockIdx.x * blockDim.x + threadIdx.x;
  if (i < n) p[i] = 0.0f;
}

// Gather entire embedded sequence to f16 once: xemb[t][b][u] = emb[ids[b][t]][u]
// Vectorized 4 elements/thread (float4 read, 4xf16 write).
__global__ void embed_gather(const float* __restrict__ emb, const int* __restrict__ ids,
                             _Float16* __restrict__ xemb) {
  size_t idx = ((size_t)blockIdx.x * blockDim.x + threadIdx.x) * 4;  // S*B*U elems
  if (idx >= (size_t)SSZ * BSZ * USZ) return;
  int u    = (int)(idx & (USZ - 1));
  int rest = (int)(idx >> 9);
  int b    = rest & (BSZ - 1);
  int t    = rest >> 7;
  int tok  = ids[b * SSZ + t];
  const float* src = emb + (size_t)tok * USZ + u;
  _Float16* dst = xemb + idx;
#pragma unroll
  for (int j = 0; j < 4; ++j) dst[j] = (_Float16)src[j];
}

// ---------------------------------------------------------------------------
// Recurrent step kernels.  Grid 32 x 256 threads = 256 waves = 256 (16x16)
// C tiles of the [128 x 512] step output.  tile = blockIdx*8 + wave; the 8
// waves of a block share one M tile (A rows hit WGP$/L2).
//
// Highway layer: h = tanh(A@Wh + bh); t = sigmoid(A@Wt + bt);
// s_out = (h - s_in)*t + s_in.
// CONCAT=1: A = [x_t | s] (K=1024); CONCAT=0: A = s (K=512).
// One A fragment feeds BOTH WMMAs; branch-free fully-unrolled K loop.
// State kept as f32 master (accumulation precision) + f16 mirror (matmul A).
// ---------------------------------------------------------------------------
template <int KTILES, bool CONCAT>
__global__ void hw_step_t(const _Float16* __restrict__ x16,
                          const _Float16* __restrict__ s_in16,
                          const float* __restrict__ s_in32,
                          float* __restrict__ s_out32,
                          _Float16* __restrict__ s_out16,
                          const _Float16* __restrict__ Whp, const _Float16* __restrict__ Wtp,
                          const float* __restrict__ bh, const float* __restrict__ bt) {
  int lane = threadIdx.x & 31;
  int wave = threadIdx.x >> 5;
  int tile = blockIdx.x * 8 + wave;   // 0..255
  int mtile = tile >> 5;              // 0..7   (128 rows / 16)
  int ntile = tile & 31;              // 0..31  (512 cols / 16)
  int half = lane >> 4;
  int mrow = mtile * 16 + (lane & 15);
  const _Float16* xrow = CONCAT ? (x16 + (size_t)mrow * USZ) : nullptr;
  const _Float16* srow = s_in16 + (size_t)mrow * USZ;
  v8f hacc = {};
  v8f tacc = {};
#pragma unroll
  for (int kt = 0; kt < KTILES; ++kt) {
    v16h a;
    if (CONCAT) {
      if (kt < 16) a = load_a_f16(xrow, kt * 32, half);
      else         a = load_a_f16(srow, (kt - 16) * 32, half);
    } else {
      a = load_a_f16(srow, kt * 32, half);
    }
    const size_t fb = ((size_t)(kt * 32 + ntile)) * 512 + lane * 16;
    v16h bwh = load_b_frag(Whp + fb);
    v16h bwt = load_b_frag(Wtp + fb);
    hacc = wmma_f16(a, bwh, hacc);
    tacc = wmma_f16(a, bwt, tacc);
  }
  int nc = ntile * 16 + (lane & 15);
  float bhv = bh[nc], btv = bt[nc];
#pragma unroll
  for (int i = 0; i < 8; ++i) {
    int mr = mtile * 16 + half * 8 + i;
    float sold = s_in32[(size_t)mr * USZ + nc];
    float h  = tanhf(hacc[i] + bhv);
    float tg = 1.0f / (1.0f + __expf(-(tacc[i] + btv)));
    float sn = (h - sold) * tg + sold;
    s_out32[(size_t)mr * USZ + nc] = sn;
    s_out16[(size_t)mr * USZ + nc] = (_Float16)sn;
  }
}

// Projection: flat[b*S + t][:] = s @ Wp + bp   (stored f16 for the NCE GEMM)
__global__ void proj_step(const _Float16* __restrict__ s_in16, const _Float16* __restrict__ Wpp,
                          const float* __restrict__ bp, _Float16* __restrict__ flat, int t) {
  int lane = threadIdx.x & 31;
  int wave = threadIdx.x >> 5;
  int tile = blockIdx.x * 8 + wave;
  int mtile = tile >> 5;
  int ntile = tile & 31;
  int half = lane >> 4;
  int mrow = mtile * 16 + (lane & 15);
  const _Float16* srow = s_in16 + (size_t)mrow * USZ;
  v8f acc = {};
#pragma unroll
  for (int kt = 0; kt < 16; ++kt) {
    v16h a = load_a_f16(srow, kt * 32, half);
    v16h b = load_b_frag(Wpp + ((size_t)(kt * 32 + ntile)) * 512 + lane * 16);
    acc = wmma_f16(a, b, acc);
  }
  int nc = ntile * 16 + (lane & 15);
  float bpv = bp[nc];
#pragma unroll
  for (int i = 0; i < 8; ++i) {
    int mr = mtile * 16 + half * 8 + i;                 // batch index
    flat[(size_t)(mr * SSZ + t) * USZ + nc] = (_Float16)(acc[i] + bpv);
  }
}

// ---------------------------------------------------------------------------
// NCE sampled-logits GEMM:  [32768 x 512] @ [512 x 64] -> softplus-summed
// per 16x16 tile into partials[tile].  8192 tiles, 1024 blocks x 8 waves.
// ---------------------------------------------------------------------------
__global__ void nce_gemm(const _Float16* __restrict__ flat, const _Float16* __restrict__ WnSp,
                         const float* __restrict__ colbias, float* __restrict__ partials) {
  int lane = threadIdx.x & 31;
  int wave = threadIdx.x >> 5;
  int tile = blockIdx.x * 8 + wave;   // 0..8191
  int rt = tile >> 2;                 // row tile 0..2047
  int ct = tile & 3;                  // col tile 0..3
  int half = lane >> 4;
  const _Float16* arow = flat + (size_t)(rt * 16 + (lane & 15)) * USZ;
  v8f c = {};
#pragma unroll
  for (int kt = 0; kt < 16; ++kt) {
    v16h a = load_a_f16(arow, kt * 32, half);
    v16h b = load_b_frag(WnSp + ((size_t)(kt * 4 + ct)) * 512 + lane * 16);
    c = wmma_f16(a, b, c);
  }
  float cb = colbias[ct * 16 + (lane & 15)];
  float s = 0.0f;
#pragma unroll
  for (int i = 0; i < 8; ++i) s += softplusf(c[i] + cb);
  for (int off = 16; off; off >>= 1) s += __shfl_xor(s, off, 32);
  if (lane == 0) partials[tile] = s;
}

// True-class logit: one wave per flat row, gathered column dot of softmax_w.
__global__ void true_logit_k(const _Float16* __restrict__ flat, const int* __restrict__ targets,
                             const float* __restrict__ sw, const float* __restrict__ sb,
                             float* __restrict__ out) {
  int lane = threadIdx.x & 31;
  int wave = threadIdx.x >> 5;
  int r = blockIdx.x * 8 + wave;      // 0..32767
  int lab = targets[r];
  const _Float16* fr = flat + (size_t)r * USZ;
  const float* wc = sw + lab;         // column of softmax_w, stride V
  float acc = 0.0f;
#pragma unroll 4
  for (int u = lane; u < USZ; u += 32) acc += (float)fr[u] * wc[(size_t)u * VSZ];
  for (int off = 16; off; off >>= 1) acc += __shfl_xor(acc, off, 32);
  if (lane == 0) {
    float tl = acc + sb[lab] - log_q_dev(lab);
    out[r] = softplusf(-tl);
  }
}

// Fixed-order final reduction -> deterministic scalar loss.
__global__ void final_reduce(const float* __restrict__ p, int n, float* __restrict__ out) {
  __shared__ float sm[256];
  float acc = 0.0f;
  for (int i = threadIdx.x; i < n; i += 256) acc += p[i];
  sm[threadIdx.x] = acc;
  __syncthreads();
  for (int s = 128; s; s >>= 1) {
    if (threadIdx.x < (unsigned)s) sm[threadIdx.x] += sm[threadIdx.x + s];
    __syncthreads();
  }
  if (threadIdx.x == 0) out[0] = sm[0] * (1.0f / ((float)BSZ * (float)SSZ));
}

// ---------------------------------------------------------------------------
extern "C" void kernel_launch(void* const* d_in, const int* in_sizes, int n_in,
                              void* d_out, int out_size, void* d_ws, size_t ws_size,
                              hipStream_t stream) {
  (void)in_sizes; (void)n_in; (void)out_size; (void)ws_size;
  const int*   input_data = (const int*)d_in[0];
  const int*   targets    = (const int*)d_in[1];
  const int*   sampled    = (const int*)d_in[2];
  const float* emb        = (const float*)d_in[3];
  const float* Wh0 = (const float*)d_in[4];  const float* bh0 = (const float*)d_in[5];
  const float* Wt0 = (const float*)d_in[6];  const float* bt0 = (const float*)d_in[7];
  const float* Wh1 = (const float*)d_in[8];  const float* bh1 = (const float*)d_in[9];
  const float* Wt1 = (const float*)d_in[10]; const float* bt1 = (const float*)d_in[11];
  const float* Wp  = (const float*)d_in[12]; const float* bp  = (const float*)d_in[13];
  const float* sw  = (const float*)d_in[14]; const float* sb  = (const float*)d_in[15];

  char* ws = (char*)d_ws;
  size_t off = 0;
  auto walloc = [&](size_t bytes) -> char* {
    char* p = ws + off;
    off = (off + bytes + 255) & ~(size_t)255;
    return p;
  };
  _Float16* flat     = (_Float16*)walloc((size_t)BSZ * SSZ * USZ * 2);  // 32 MB
  _Float16* xemb     = (_Float16*)walloc((size_t)SSZ * BSZ * USZ * 2);  // 32 MB
  float*    stateA32 = (float*)walloc((size_t)BSZ * USZ * 4);
  float*    stateB32 = (float*)walloc((size_t)BSZ * USZ * 4);
  _Float16* stateA16 = (_Float16*)walloc((size_t)BSZ * USZ * 2);
  _Float16* stateB16 = (_Float16*)walloc((size_t)BSZ * USZ * 2);
  _Float16* Wh0p     = (_Float16*)walloc((size_t)2 * USZ * USZ * 2);
  _Float16* Wt0p     = (_Float16*)walloc((size_t)2 * USZ * USZ * 2);
  _Float16* Wh1p     = (_Float16*)walloc((size_t)USZ * USZ * 2);
  _Float16* Wt1p     = (_Float16*)walloc((size_t)USZ * USZ * 2);
  _Float16* Wpp      = (_Float16*)walloc((size_t)USZ * USZ * 2);
  _Float16* WnSp     = (_Float16*)walloc((size_t)USZ * NSS * 2);
  float*    cbias    = (float*)walloc(NSS * 4);
  float*    partials = (float*)walloc((size_t)(8192 + 32768) * 4);

  // One-time packing + gathers + state init (all parallel)
  embed_gather<<<(int)(((size_t)SSZ * BSZ * USZ / 4 + 255) / 256), 256, 0, stream>>>(
      emb, input_data, xemb);
  pack_weight<<<(2 * USZ * USZ + 255) / 256, 256, 0, stream>>>(Wh0, Wh0p, 2 * USZ, USZ);
  pack_weight<<<(2 * USZ * USZ + 255) / 256, 256, 0, stream>>>(Wt0, Wt0p, 2 * USZ, USZ);
  pack_weight<<<(USZ * USZ + 255) / 256, 256, 0, stream>>>(Wh1, Wh1p, USZ, USZ);
  pack_weight<<<(USZ * USZ + 255) / 256, 256, 0, stream>>>(Wt1, Wt1p, USZ, USZ);
  pack_weight<<<(USZ * USZ + 255) / 256, 256, 0, stream>>>(Wp, Wpp, USZ, USZ);
  pack_wns<<<(USZ * NSS + 255) / 256, 256, 0, stream>>>(sw, sampled, WnSp);
  colbias_k<<<1, 64, 0, stream>>>(sb, sampled, cbias);
  zero_f32<<<(BSZ * USZ + 255) / 256, 256, 0, stream>>>(stateA32, BSZ * USZ);
  zero_f32<<<(BSZ * USZ / 2 + 255) / 256, 256, 0, stream>>>((float*)stateA16, BSZ * USZ / 2);

  // Serial recurrent scan: state ping-pongs A -> B -> A each step.
  for (int t = 0; t < SSZ; ++t) {
    hw_step_t<32, true><<<32, 256, 0, stream>>>(
        xemb + (size_t)t * BSZ * USZ, stateA16, stateA32, stateB32, stateB16,
        Wh0p, Wt0p, bh0, bt0);
    hw_step_t<16, false><<<32, 256, 0, stream>>>(
        nullptr, stateB16, stateB32, stateA32, stateA16,
        Wh1p, Wt1p, bh1, bt1);
    proj_step<<<32, 256, 0, stream>>>(stateA16, Wpp, bp, flat, t);
  }

  // NCE loss
  nce_gemm<<<1024, 256, 0, stream>>>(flat, WnSp, cbias, partials);
  true_logit_k<<<4096, 256, 0, stream>>>(flat, targets, sw, sb, partials + 8192);
  final_reduce<<<1, 256, 0, stream>>>(partials, 8192 + 32768, (float*)d_out);
}
